// Actor_68375879352861
// MI455X (gfx1250) — matrix operand order repaired and checked by reading, not compiled, not run on hardware
//
#include <hip/hip_runtime.h>

// Problem constants (match reference): F_IN=64, H=128, K=4 -> concat K-dim 256
#define F_IN 64
#define HDIM 128
#define KORD 4
#define KTOT (F_IN * KORD)   // 256

typedef __attribute__((ext_vector_type(16))) _Float16 v16h;
typedef __attribute__((ext_vector_type(8)))  _Float16 v8h;
typedef __attribute__((ext_vector_type(8)))  float    v8f;

// ---------------------------------------------------------------- utilities
__global__ void fill_f32_kernel(float* __restrict__ p, float v, int n) {
  int i = blockIdx.x * blockDim.x + threadIdx.x;
  if (i < n) p[i] = v;
}

__global__ void init_out_kernel(float* __restrict__ out,
                                const float* __restrict__ final_b, int n) {
  int i = blockIdx.x * blockDim.x + threadIdx.x;
  if (i < n) out[i] = final_b[0];
}

// ---------------------------------------------------------------- normalization
__global__ void deg_kernel(const int* __restrict__ rows,
                           float* __restrict__ deg, int e) {
  int i = blockIdx.x * blockDim.x + threadIdx.x;
  if (i < e) atomicAdd(&deg[rows[i]], 1.0f);
}

__global__ void dinv_kernel(float* __restrict__ deg, int n) {
  int i = blockIdx.x * blockDim.x + threadIdx.x;
  if (i < n) {
    float d = deg[i];
    deg[i] = (d > 0.0f) ? rsqrtf(d) : 0.0f;   // deg>=1 when >0
  }
}

__global__ void ew_kernel(const int* __restrict__ rows,
                          const int* __restrict__ cols,
                          const float* __restrict__ dinv,
                          float* __restrict__ ew, int e) {
  int i = blockIdx.x * blockDim.x + threadIdx.x;
  if (i < e) ew[i] = -dinv[rows[i]] * dinv[cols[i]];
}

// ---------------------------------------------------------------- SpMM: S[row] += ew * Z[col]
// 16 threads per edge, each handles 4 consecutive features via one float4 gather.
__global__ void spmm_kernel(const int* __restrict__ rows,
                            const int* __restrict__ cols,
                            const float* __restrict__ ew,
                            const float* __restrict__ Z,   // [N, 64] f32
                            float* __restrict__ S,         // [N, 64] f32 (pre-zeroed)
                            int nedges) {
  int gid = blockIdx.x * blockDim.x + threadIdx.x;
  int e = gid >> 4;
  int q = gid & 15;
  if (e >= nedges) return;
  int r = rows[e];
  int c = cols[e];
  float w = ew[e];
  float4 z = ((const float4*)Z)[(size_t)c * 16 + q];
  float* dst = S + (size_t)r * F_IN + q * 4;
  atomicAdd(dst + 0, w * z.x);
  atomicAdd(dst + 1, w * z.y);
  atomicAdd(dst + 2, w * z.z);
  atomicAdd(dst + 3, w * z.w);
}

// ---------------------------------------------------------------- Chebyshev combine + f16 convert
// T = alpha*T + beta*Tprev (in place, f32), and Xh[:, blk*64 : blk*64+64] = (f16)T
__global__ void combine_convert_kernel(float* __restrict__ T,
                                       const float* __restrict__ Tprev,
                                       float alpha, float beta,
                                       _Float16* __restrict__ Xh, int blk, int cnt) {
  int i = blockIdx.x * blockDim.x + threadIdx.x;
  if (i >= cnt) return;
  float t = alpha * T[i] + beta * Tprev[i];
  T[i] = t;
  int nrow = i >> 6;            // /64
  int f    = i & 63;
  Xh[(size_t)nrow * KTOT + blk * F_IN + f] = (_Float16)t;
}

// Xh block 0 straight from the (immutable) input features
__global__ void conv_x0_kernel(const float* __restrict__ X,
                               _Float16* __restrict__ Xh, int cnt) {
  int i = blockIdx.x * blockDim.x + threadIdx.x;
  if (i >= cnt) return;
  int nrow = i >> 6;
  int f    = i & 63;
  Xh[(size_t)nrow * KTOT + f] = (_Float16)X[i];
}

// cheb_w[K,64,128] (flat: k*128+h over k in 0..255) -> Wth[h][k] f16 (column-major weights)
__global__ void conv_w_kernel(const float* __restrict__ cw,
                              _Float16* __restrict__ Wth) {
  int i = blockIdx.x * blockDim.x + threadIdx.x;   // 0 .. 128*256-1
  if (i >= HDIM * KTOT) return;
  int h = i >> 8;     // 0..127
  int k = i & 255;    // 0..255
  Wth[(size_t)h * KTOT + k] = (_Float16)cw[(size_t)k * HDIM + h];
}

// ---------------------------------------------------------------- fused WMMA GEMM + tanh + final dot
// out[m] += sum_n tanh( (Xh[m,:] . Wth[n,:]) + cheb_b[n] ) * final_w[n]
// Block = 8 waves; wave w owns N-tile [w*16, w*16+16); block b owns M-tile [b*16, b*16+16).
__global__ void __launch_bounds__(256)
gemm_tanh_dot_kernel(const _Float16* __restrict__ Xh,    // [N, 256]
                     const _Float16* __restrict__ Wth,   // [128, 256]
                     const float* __restrict__ cheb_b,   // [128]
                     const float* __restrict__ final_w,  // [128]
                     float* __restrict__ out,            // [N], pre-seeded with final_b
                     int n_rows) {
  const int lane  = threadIdx.x & 31;
  const int wave  = threadIdx.x >> 5;       // 0..7  -> nbase = wave*16
  const int mbase = blockIdx.x * 16;
  const int nbase = wave * 16;
  const int ln    = lane & 15;
  const int hi    = lane >> 4;              // half-wave select

  int m = mbase + ln;
  if (m >= n_rows) m = n_rows - 1;          // clamp loads (stores are guarded)
  const int n = nbase + ln;

  const _Float16* arow = Xh  + (size_t)m * KTOT;
  const _Float16* bcol = Wth + (size_t)n * KTOT;

  v8f c = {};
  #pragma unroll
  for (int k0 = 0; k0 < KTOT; k0 += 32) {
    // 16-bit A 16x32 layout: lanes 0-15 hold K {0..7, 16..23}, lanes 16-31 hold K {8..15, 24..31}
    union { v16h v; v8h h[2]; } a;
    a.h[0] = *(const v8h*)(arow + k0 + hi * 8);
    a.h[1] = *(const v8h*)(arow + k0 + hi * 8 + 16);
    // 16-bit B 32x16 layout: lane = column, same K order as A: lanes 0-15 K {0..7, 16..23}, lanes 16-31 K {8..15, 24..31}
    union { v16h v; v8h h[2]; } bu;
    bu.h[0] = *(const v8h*)(bcol + k0 + hi * 8);
    bu.h[1] = *(const v8h*)(bcol + k0 + hi * 8 + 16);
    v16h b = bu.v;
    c = __builtin_amdgcn_wmma_f32_16x16x32_f16(false, a.v, false, b,
                                               (short)0, c, false, false);
  }

  // C/D layout: lane (ln, hi) vgpr r -> element (M = r + 8*hi, N = ln)
  const float bias = cheb_b[n];
  const float fw   = final_w[n];
  #pragma unroll
  for (int r = 0; r < 8; ++r) {
    int mrow = mbase + r + hi * 8;
    float v = tanhf(c[r] + bias) * fw;
    // reduce the 16 columns of this tile across the half-wave (masks < 16 never cross halves)
    v += __shfl_xor(v, 8, 32);
    v += __shfl_xor(v, 4, 32);
    v += __shfl_xor(v, 2, 32);
    v += __shfl_xor(v, 1, 32);
    if (ln == 0 && mrow < n_rows) atomicAdd(&out[mrow], v);
  }
}

// ---------------------------------------------------------------- host-side launch
extern "C" void kernel_launch(void* const* d_in, const int* in_sizes, int n_in,
                              void* d_out, int out_size, void* d_ws, size_t ws_size,
                              hipStream_t stream) {
  (void)n_in; (void)out_size; (void)ws_size;

  const float* features = (const float*)d_in[0];   // [N,64]
  const int*   eidx     = (const int*)  d_in[1];   // [2,E]
  const float* cheb_w   = (const float*)d_in[2];   // [4,64,128]
  const float* cheb_b   = (const float*)d_in[3];   // [128]
  const float* final_w  = (const float*)d_in[4];   // [128]
  const float* final_b  = (const float*)d_in[5];   // [1]
  float*       out      = (float*)d_out;           // [N]

  const int N = in_sizes[0] / F_IN;
  const int E = in_sizes[1] / 2;
  const int* rows = eidx;
  const int* cols = eidx + E;

  // carve scratch
  char* ws = (char*)d_ws;
  size_t off = 0;
  auto carve = [&](size_t bytes) -> char* {
    off = (off + 255) & ~(size_t)255;
    char* p = ws + off;
    off += bytes;
    return p;
  };
  float*    deg = (float*)   carve((size_t)N * 4);             // deg -> dinv in place
  float*    ew  = (float*)   carve((size_t)E * 4);
  float*    T1  = (float*)   carve((size_t)N * F_IN * 4);
  float*    T2  = (float*)   carve((size_t)N * F_IN * 4);
  float*    T3  = (float*)   carve((size_t)N * F_IN * 4);
  _Float16* Xh  = (_Float16*)carve((size_t)N * KTOT * 2);      // [N,256] f16
  _Float16* Wth = (_Float16*)carve((size_t)HDIM * KTOT * 2);   // [128,256] f16

  const int B = 256;
  auto cdiv = [](long long a, long long b) { return (int)((a + b - 1) / b); };
  const int NF = N * F_IN;
  const long long spmm_threads = (long long)E * 16;

  // normalization
  fill_f32_kernel<<<cdiv(N, B), B, 0, stream>>>(deg, 0.0f, N);
  deg_kernel     <<<cdiv(E, B), B, 0, stream>>>(rows, deg, E);
  dinv_kernel    <<<cdiv(N, B), B, 0, stream>>>(deg, N);
  ew_kernel      <<<cdiv(E, B), B, 0, stream>>>(rows, cols, deg, ew, E);

  // T0 = features -> Xh block 0
  conv_x0_kernel<<<cdiv(NF, B), B, 0, stream>>>(features, Xh, NF);

  // T1 = L @ T0
  fill_f32_kernel<<<cdiv(NF, B), B, 0, stream>>>(T1, 0.0f, NF);
  spmm_kernel<<<cdiv(spmm_threads, B), B, 0, stream>>>(rows, cols, ew, features, T1, E);
  combine_convert_kernel<<<cdiv(NF, B), B, 0, stream>>>(T1, features, 1.0f, 0.0f, Xh, 1, NF);

  // T2 = 2*(L @ T1) - T0
  fill_f32_kernel<<<cdiv(NF, B), B, 0, stream>>>(T2, 0.0f, NF);
  spmm_kernel<<<cdiv(spmm_threads, B), B, 0, stream>>>(rows, cols, ew, T1, T2, E);
  combine_convert_kernel<<<cdiv(NF, B), B, 0, stream>>>(T2, features, 2.0f, -1.0f, Xh, 2, NF);

  // T3 = 2*(L @ T2) - T1
  fill_f32_kernel<<<cdiv(NF, B), B, 0, stream>>>(T3, 0.0f, NF);
  spmm_kernel<<<cdiv(spmm_threads, B), B, 0, stream>>>(rows, cols, ew, T2, T3, E);
  combine_convert_kernel<<<cdiv(NF, B), B, 0, stream>>>(T3, T1, 2.0f, -1.0f, Xh, 3, NF);

  // weights -> f16 transposed; seed output with final bias
  conv_w_kernel<<<cdiv(HDIM * KTOT, B), B, 0, stream>>>(cheb_w, Wth);
  init_out_kernel<<<cdiv(N, B), B, 0, stream>>>(out, final_b, N);

  // fused WMMA GEMM + tanh + final projection
  gemm_tanh_dot_kernel<<<cdiv(N, 16), 256, 0, stream>>>(Xh, Wth, cheb_b, final_w, out, N);
}
